// NestedTensorBlock_13932873908746
// MI455X (gfx1250) — matrix operand order repaired and checked by reading, not compiled, not run on hardware
//
#include <hip/hip_runtime.h>
#include <hip/hip_bf16.h>

typedef __attribute__((ext_vector_type(16))) _Float16 v16h;
typedef __attribute__((ext_vector_type(8)))  _Float16 v8h;
typedef __attribute__((ext_vector_type(8)))  float    v8f;

__device__ __forceinline__ v16h make_v16h(v8h lo, v8h hi) {
  v16h r;
#pragma unroll
  for (int i = 0; i < 8; ++i) { r[i] = lo[i]; r[i + 8] = hi[i]; }
  return r;
}

__device__ __forceinline__ v8f wmma_f16(v16h a, v16h b, v8f c) {
  // D = A(16x32) * B(32x16) + C, f32 accumulate
  return __builtin_amdgcn_wmma_f32_16x16x32_f16(
      /*neg_a=*/false, a, /*neg_b=*/false, b,
      /*c_mod=*/(short)0, c, /*reuse_a=*/false, /*reuse_b=*/false);
}

// CDNA5 async global->LDS copy (ASYNCcnt-tracked, bypasses VGPRs).
// lds_off: byte offset of destination within the workgroup LDS allocation.
__device__ __forceinline__ void async_ld_b128(unsigned lds_off, const _Float16* g) {
  asm volatile("global_load_async_to_lds_b128 %0, %1, off"
               :: "v"(lds_off), "v"(g)
               : "memory");
}
__device__ __forceinline__ void wait_async0() {
  asm volatile("s_wait_asynccnt 0x0" ::: "memory");
}
__device__ __forceinline__ unsigned lds_off_of(const void* p) {
  // generic pointer to LDS: low 32 bits are the LDS byte address
  return (unsigned)(size_t)p;
}

// ---------------------------------------------------------------------------
// fp32 -> f16 weight conversion
// ---------------------------------------------------------------------------
__global__ void cvt_kernel(const float* __restrict__ s, _Float16* __restrict__ d, int n) {
  int i = blockIdx.x * 256 + threadIdx.x;
  if (i < n) d[i] = (_Float16)s[i];
}

// ---------------------------------------------------------------------------
// LayerNorm: one block per token, fp32 stats, f16 output
// srcB == nullptr -> single source; else token t>=T0 reads srcB.
// ---------------------------------------------------------------------------
__global__ __launch_bounds__(256) void ln_kernel(
    const float* __restrict__ srcA, const float* __restrict__ srcB, int T0,
    const float* __restrict__ w, const float* __restrict__ bb,
    _Float16* __restrict__ out) {
  const int t = blockIdx.x;
  const float* row = (srcB != nullptr && t >= T0)
                         ? (srcB + (size_t)(t - T0) * 768)
                         : (srcA + (size_t)t * 768);
  __shared__ float r1[256], r2[256];
  float s = 0.f, s2 = 0.f;
  for (int i = threadIdx.x; i < 768; i += 256) {
    float v = row[i];
    s += v; s2 += v * v;
  }
  r1[threadIdx.x] = s; r2[threadIdx.x] = s2;
  __syncthreads();
  for (int off = 128; off > 0; off >>= 1) {
    if (threadIdx.x < off) {
      r1[threadIdx.x] += r1[threadIdx.x + off];
      r2[threadIdx.x] += r2[threadIdx.x + off];
    }
    __syncthreads();
  }
  const float mean = r1[0] * (1.0f / 768.0f);
  const float var  = r2[0] * (1.0f / 768.0f) - mean * mean;
  const float rs   = rsqrtf(var + 1e-5f);
  for (int i = threadIdx.x; i < 768; i += 256)
    out[(size_t)t * 768 + i] = (_Float16)((row[i] - mean) * rs * w[i] + bb[i]);
}

// ---------------------------------------------------------------------------
// Generic GEMM: C[M x N] = A16[M x K] * W16[N x K]^T  (both f16, f32 accum)
// Block 256 thr = 8 waves, tile 128x128x32. Wave = 64x32 (4x2 wmma tiles).
// Double-buffered LDS filled by async global->LDS DMA (ASYNCcnt).
// MODE: 0=QKV scatter, 1=proj(+bias+input residual)->f32,
//       2=fc1(+bias+gelu)->f16, 3=fc2(+bias+X1 residual)->f32
// ---------------------------------------------------------------------------
struct EpiParams {
  const float* bias;
  const float* resid;   // fc2: X1
  const float* xin1;    // proj: original x1
  const float* xin2;    // proj: original x2
  float*       out_f32;
  _Float16*    out_f16;
  _Float16 *q, *k, *v;  // qkv scatter targets
  int T0, N0, N1;       // segment info (tokens in seg0, seq lens)
};

template <int MODE>
__global__ __launch_bounds__(256) void gemm_ws(
    const _Float16* __restrict__ A, const _Float16* __restrict__ W,
    int Mtot, int K, EpiParams ep) {
  __shared__ _Float16 As[2][128][32];  // [buf][row][k]
  __shared__ _Float16 Bs[2][128][32];  // [buf][n][k] (transposed)

  const int tid  = threadIdx.x;
  const int lane = tid & 31;
  const int wid  = tid >> 5;
  const int wm   = wid & 1;   // 0..1 : 64-row half
  const int wn   = wid >> 1;  // 0..3 : 32-col strip
  const int m0   = blockIdx.x * 128;
  const int n0   = blockIdx.y * 128;

  v8f acc[4][2];
#pragma unroll
  for (int i = 0; i < 4; ++i)
#pragma unroll
    for (int j = 0; j < 2; ++j) acc[i][j] = (v8f){};

  const int  ldr  = tid >> 1;          // 0..127
  const int  ldc  = (tid & 1) * 16;    // 0 or 16
  const long arow = m0 + ldr;
  const long brow = n0 + ldr;

  const int r15 = lane & 15;
  const int kbA = (lane < 16) ? 0 : 8;
  const int kbB = (lane < 16) ? 0 : 8;

  // per-thread async staging of one 16-half (32B) strip of A and B
  auto stage = [&](int k0, int buf) {
    const _Float16* pb = W + (size_t)brow * K + k0 + ldc;
    const unsigned  lb = lds_off_of(&Bs[buf][ldr][ldc]);
    async_ld_b128(lb, pb);
    async_ld_b128(lb + 16, pb + 8);
    if (arow < Mtot) {  // skipped lanes leave stale LDS; only feeds discarded C rows
      const _Float16* pa = A + (size_t)arow * K + k0 + ldc;
      const unsigned  la = lds_off_of(&As[buf][ldr][ldc]);
      async_ld_b128(la, pa);
      async_ld_b128(la + 16, pa + 8);
    }
  };

  const int T = K >> 5;  // K/32, exact
  stage(0, 0);
  for (int t = 0; t < T; ++t) {
    const int cur = t & 1;
    wait_async0();       // own async copies for tile t complete
    __syncthreads();     // all waves' copies complete; all done reading buf 1-cur
    if (t + 1 < T) stage((t + 1) * 32, 1 - cur);  // DMA next tile during compute

    v16h af[4], bf[2];
#pragma unroll
    for (int i = 0; i < 4; ++i) {
      const int row = wm * 64 + i * 16 + r15;
      af[i] = make_v16h(*(const v8h*)&As[cur][row][kbA],
                        *(const v8h*)&As[cur][row][kbA + 16]);
    }
#pragma unroll
    for (int j = 0; j < 2; ++j) {
      const int col = wn * 32 + j * 16 + r15;
      bf[j] = make_v16h(*(const v8h*)&Bs[cur][col][kbB],
                        *(const v8h*)&Bs[cur][col][kbB + 16]);
    }
#pragma unroll
    for (int i = 0; i < 4; ++i)
#pragma unroll
      for (int j = 0; j < 2; ++j) acc[i][j] = wmma_f16(af[i], bf[j], acc[i][j]);
  }

  // ---------------- epilogue ----------------
  const int roff = (lane < 16) ? 0 : 8;
  float bv[2];
#pragma unroll
  for (int j = 0; j < 2; ++j) {
    const int ncol = n0 + wn * 32 + j * 16 + r15;
    bv[j] = (MODE != 0) ? ep.bias[ncol] : 0.0f;
  }
#pragma unroll
  for (int i = 0; i < 4; ++i) {
#pragma unroll
    for (int rr = 0; rr < 8; ++rr) {
      const int row = m0 + wm * 64 + i * 16 + rr + roff;
      if (row >= Mtot) continue;

      if (MODE == 0) {
        // hoist token -> (segment, b, n) once per row
        int segbase, Nseg, local;
        if (row < ep.T0) { segbase = 0;     Nseg = ep.N0; local = row; }
        else             { segbase = ep.T0; Nseg = ep.N1; local = row - ep.T0; }
        const int    b       = local / Nseg;
        const int    n       = local - b * Nseg;
        const size_t rowbase = (size_t)segbase * 768 +
                               ((size_t)b * 12 * Nseg + n) * 64;
        const size_t hstride = (size_t)Nseg * 64;
#pragma unroll
        for (int j = 0; j < 2; ++j) {
          const int   ncol  = n0 + wn * 32 + j * 16 + r15;
          const float val   = acc[i][j][rr];
          const int   which = ncol / 768;
          const int   rem   = ncol - which * 768;
          const int   h     = rem >> 6;
          const int   d     = rem & 63;
          const size_t dst  = rowbase + (size_t)h * hstride + d;
          if (which == 0)      ep.q[dst] = (_Float16)(val * 0.125f);  // hd^-0.5
          else if (which == 1) ep.k[dst] = (_Float16)val;
          else                 ep.v[dst] = (_Float16)val;
        }
      } else if (MODE == 1) {
        const float* xr = (row < ep.T0)
                              ? (ep.xin1 + (size_t)row * 768)
                              : (ep.xin2 + (size_t)(row - ep.T0) * 768);
#pragma unroll
        for (int j = 0; j < 2; ++j) {
          const int ncol = n0 + wn * 32 + j * 16 + r15;
          ep.out_f32[(size_t)row * 768 + ncol] = acc[i][j][rr] + bv[j] + xr[ncol];
        }
      } else if (MODE == 2) {
#pragma unroll
        for (int j = 0; j < 2; ++j) {
          const int ncol = n0 + wn * 32 + j * 16 + r15;
          float g = acc[i][j][rr] + bv[j];
          g = 0.5f * g * (1.0f + erff(g * 0.70710678118f));  // exact GELU
          ep.out_f16[(size_t)row * 3072 + ncol] = (_Float16)g;
        }
      } else {
#pragma unroll
        for (int j = 0; j < 2; ++j) {
          const int ncol = n0 + wn * 32 + j * 16 + r15;
          ep.out_f32[(size_t)row * 768 + ncol] =
              acc[i][j][rr] + bv[j] + ep.resid[(size_t)row * 768 + ncol];
        }
      }
    }
  }
}

// ---------------------------------------------------------------------------
// Flash attention: block = 4 waves, each wave owns 16 query rows.
// grid = (ceil(ceil(N/16)/4), H, B). Streams 32-key tiles through LDS.
// Q pre-scaled by hd^-0.5. Writes O as f16 [token][768] (head slice).
// ---------------------------------------------------------------------------
__global__ __launch_bounds__(128) void attn_kernel(
    const _Float16* __restrict__ Q, const _Float16* __restrict__ Km,
    const _Float16* __restrict__ Vm, _Float16* __restrict__ O, int N) {
  const int b = blockIdx.z, h = blockIdx.y;
  const int tid = threadIdx.x, lane = tid & 31, w = tid >> 5;
  const int qb = blockIdx.x * 4 + w;
  const int qbase = qb * 16;

  __shared__ _Float16 Kt[32][64];       // [key][d]
  __shared__ _Float16 Vt[64][32];       // [d][key] (transposed)
  __shared__ float    Sbuf[4][16][32];  // per-wave scores
  __shared__ _Float16 Pbuf[4][16][32];  // per-wave probs (f16)
  __shared__ float    Fbuf[4][16];      // per-wave row factors

  const size_t bh = ((size_t)b * 12 + h) * (size_t)N;

  const int r15  = lane & 15;
  const int kbA  = (lane < 16) ? 0 : 8;
  const int kbB  = (lane < 16) ? 0 : 8;
  const int roff = (lane < 16) ? 0 : 8;

  // load Q fragments (A-layout) directly from global, row-clamped
  int qrow = qbase + r15;
  if (qrow >= N) qrow = N - 1;
  const _Float16* qp = Q + (bh + qrow) * 64;
  const v16h aq0 = make_v16h(*(const v8h*)(qp + kbA), *(const v8h*)(qp + kbA + 16));
  const v16h aq1 = make_v16h(*(const v8h*)(qp + 32 + kbA), *(const v8h*)(qp + 32 + kbA + 16));

  v8f oacc[4];
#pragma unroll
  for (int f = 0; f < 4; ++f) oacc[f] = (v8f){};
  float mreg = -1e30f, lreg = 0.0f;

  const int ldk = tid >> 2;        // key 0..31
  const int ldd = (tid & 3) * 16;  // d offset

  const int ntiles = (N + 31) / 32;
  for (int kt = 0; kt < ntiles; ++kt) {
    __syncthreads();
    {
      const int kg = kt * 32 + ldk;
      v8h k0 = (v8h){}, k1 = (v8h){}, v0 = (v8h){}, v1 = (v8h){};
      if (kg < N) {
        const _Float16* kp = Km + (bh + kg) * 64 + ldd;
        k0 = *(const v8h*)kp;  k1 = *(const v8h*)(kp + 8);
        const _Float16* vp = Vm + (bh + kg) * 64 + ldd;
        v0 = *(const v8h*)vp;  v1 = *(const v8h*)(vp + 8);
      }
      *(v8h*)&Kt[ldk][ldd]     = k0;
      *(v8h*)&Kt[ldk][ldd + 8] = k1;
#pragma unroll
      for (int jj = 0; jj < 8; ++jj) {
        Vt[ldd + jj][ldk]     = v0[jj];
        Vt[ldd + 8 + jj][ldk] = v1[jj];
      }
    }
    __syncthreads();

    // S = Q * K^T : two 16-key subtiles x two 32-d halves
    v8f s0 = (v8f){}, s1 = (v8f){};
    {
      const int key0 = r15, key1 = 16 + r15;
      v16h bf;
      bf = make_v16h(*(const v8h*)&Kt[key0][kbB],      *(const v8h*)&Kt[key0][kbB + 16]);
      s0 = wmma_f16(aq0, bf, s0);
      bf = make_v16h(*(const v8h*)&Kt[key0][32 + kbB], *(const v8h*)&Kt[key0][32 + kbB + 16]);
      s0 = wmma_f16(aq1, bf, s0);
      bf = make_v16h(*(const v8h*)&Kt[key1][kbB],      *(const v8h*)&Kt[key1][kbB + 16]);
      s1 = wmma_f16(aq0, bf, s1);
      bf = make_v16h(*(const v8h*)&Kt[key1][32 + kbB], *(const v8h*)&Kt[key1][32 + kbB + 16]);
      s1 = wmma_f16(aq1, bf, s1);
    }
    // spill S (C-layout) to wave-local LDS
#pragma unroll
    for (int rr = 0; rr < 8; ++rr) {
      Sbuf[w][rr + roff][r15]      = s0[rr];
      Sbuf[w][rr + roff][r15 + 16] = s1[rr];
    }
    // online softmax: lanes 0..15 each own one row (wave-internal LDS is in-order)
    if (lane < 16) {
      const int row = lane;
      float mmax = mreg;
      for (int k2 = 0; k2 < 32; ++k2)
        if (kt * 32 + k2 < N) mmax = fmaxf(mmax, Sbuf[w][row][k2]);
      const float c = __expf(mreg - mmax);
      float lsum = 0.0f;
      for (int k2 = 0; k2 < 32; ++k2) {
        float p = 0.0f;
        if (kt * 32 + k2 < N) {
          p = __expf(Sbuf[w][row][k2] - mmax);
          lsum += p;
        }
        Pbuf[w][row][k2] = (_Float16)p;
      }
      mreg = mmax;
      lreg = lreg * c + lsum;
      Fbuf[w][row] = c;
    }
    // rescale running output accumulators
    float fr[8];
#pragma unroll
    for (int rr = 0; rr < 8; ++rr) fr[rr] = Fbuf[w][rr + roff];
#pragma unroll
    for (int f = 0; f < 4; ++f)
#pragma unroll
      for (int rr = 0; rr < 8; ++rr) oacc[f][rr] *= fr[rr];

    // P as A-fragment; O += P(16x32) * V(32x64)
    const v16h pf = make_v16h(*(const v8h*)&Pbuf[w][r15][kbA],
                              *(const v8h*)&Pbuf[w][r15][kbA + 16]);
#pragma unroll
    for (int f = 0; f < 4; ++f) {
      const int dcol = f * 16 + r15;
      const v16h vf = make_v16h(*(const v8h*)&Vt[dcol][kbB],
                                *(const v8h*)&Vt[dcol][kbB + 16]);
      oacc[f] = wmma_f16(pf, vf, oacc[f]);
    }
  }

  // finalize: divide by l, store f16 head slice
  if (lane < 16) Fbuf[w][lane] = (lreg > 0.0f) ? (1.0f / lreg) : 0.0f;
  float inv[8];
#pragma unroll
  for (int rr = 0; rr < 8; ++rr) inv[rr] = Fbuf[w][rr + roff];
#pragma unroll
  for (int f = 0; f < 4; ++f) {
    const int d = f * 16 + r15;
#pragma unroll
    for (int rr = 0; rr < 8; ++rr) {
      const int row = qbase + rr + roff;
      if (row < N)
        O[((size_t)b * N + row) * 768 + h * 64 + d] = (_Float16)(oacc[f][rr] * inv[rr]);
    }
  }
}

// ---------------------------------------------------------------------------
// Host-side orchestration
// ---------------------------------------------------------------------------
extern "C" void kernel_launch(void* const* d_in, const int* in_sizes, int n_in,
                              void* d_out, int out_size, void* d_ws, size_t ws_size,
                              hipStream_t stream) {
  const float* x1     = (const float*)d_in[0];
  const float* x2     = (const float*)d_in[1];
  const float* ln1_w  = (const float*)d_in[2];
  const float* ln1_b  = (const float*)d_in[3];
  const float* qkv_w  = (const float*)d_in[4];
  const float* proj_w = (const float*)d_in[5];
  const float* proj_b = (const float*)d_in[6];
  const float* ln2_w  = (const float*)d_in[7];
  const float* ln2_b  = (const float*)d_in[8];
  const float* fc1_w  = (const float*)d_in[9];
  const float* fc1_b  = (const float*)d_in[10];
  const float* fc2_w  = (const float*)d_in[11];
  const float* fc2_b  = (const float*)d_in[12];

  constexpr int D = 768, Dff = 3072, Hh = 12, B = 8, N0 = 1370, N1 = 257;
  constexpr int T0 = B * N0;          // 10960 tokens (seg0)
  constexpr int Ttot = T0 + B * N1;   // 13016 tokens

  // workspace layout
  char* ws = (char*)d_ws;
  auto alloc = [&](size_t bytes) -> char* {
    char* p = ws;
    ws += (bytes + 255) & ~(size_t)255;
    return p;
  };
  _Float16* Wqkv  = (_Float16*)alloc((size_t)3 * D * D * 2);
  _Float16* Wproj = (_Float16*)alloc((size_t)D * D * 2);
  _Float16* Wfc1  = (_Float16*)alloc((size_t)Dff * D * 2);
  _Float16* Wfc2  = (_Float16*)alloc((size_t)D * Dff * 2);
  _Float16* A16   = (_Float16*)alloc((size_t)Ttot * D * 2);   // LN out / attn out (reused)
  float*    X1    = (float*)alloc((size_t)Ttot * D * 4);      // post-attn residual
  size_t qkv_bytes = (size_t)3 * Ttot * D * 2;
  size_t h_bytes   = (size_t)Ttot * Dff * 2;
  char* big = alloc(qkv_bytes > h_bytes ? qkv_bytes : h_bytes);
  _Float16* Qb = (_Float16*)big;
  _Float16* Kb = Qb + (size_t)Ttot * D;
  _Float16* Vb = Kb + (size_t)Ttot * D;
  _Float16* Hb = (_Float16*)big;  // fc1 activations (reuses Q/K/V region)

  // 1) weights -> f16
  cvt_kernel<<<(3 * D * D + 255) / 256, 256, 0, stream>>>(qkv_w, Wqkv, 3 * D * D);
  cvt_kernel<<<(D * D + 255) / 256, 256, 0, stream>>>(proj_w, Wproj, D * D);
  cvt_kernel<<<(Dff * D + 255) / 256, 256, 0, stream>>>(fc1_w, Wfc1, Dff * D);
  cvt_kernel<<<(D * Dff + 255) / 256, 256, 0, stream>>>(fc2_w, Wfc2, D * Dff);

  // 2) LN1 over concatenated x1|x2
  ln_kernel<<<Ttot, 256, 0, stream>>>(x1, x2, T0, ln1_w, ln1_b, A16);

  // 3) QKV GEMM with scatter
  {
    EpiParams ep{};
    ep.q = Qb; ep.k = Kb; ep.v = Vb;
    ep.T0 = T0; ep.N0 = N0; ep.N1 = N1;
    gemm_ws<0><<<dim3((Ttot + 127) / 128, (3 * D) / 128), 256, 0, stream>>>(
        A16, Wqkv, Ttot, D, ep);
  }

  // 4) attention (per segment), writes into A16 (token-major f16)
  {
    const int nq0 = (N0 + 15) / 16;
    attn_kernel<<<dim3((nq0 + 3) / 4, Hh, B), 128, 0, stream>>>(Qb, Kb, Vb, A16, N0);
    const int nq1 = (N1 + 15) / 16;
    attn_kernel<<<dim3((nq1 + 3) / 4, Hh, B), 128, 0, stream>>>(
        Qb + (size_t)T0 * D, Kb + (size_t)T0 * D, Vb + (size_t)T0 * D,
        A16 + (size_t)T0 * D, N1);
  }

  // 5) proj GEMM + bias + input residual -> X1 (f32)
  {
    EpiParams ep{};
    ep.bias = proj_b; ep.xin1 = x1; ep.xin2 = x2; ep.T0 = T0; ep.out_f32 = X1;
    gemm_ws<1><<<dim3((Ttot + 127) / 128, D / 128), 256, 0, stream>>>(
        A16, Wproj, Ttot, D, ep);
  }

  // 6) LN2 over X1 -> A16
  ln_kernel<<<Ttot, 256, 0, stream>>>(X1, nullptr, 0, ln2_w, ln2_b, A16);

  // 7) FC1 + bias + GELU -> Hb (f16)
  {
    EpiParams ep{};
    ep.bias = fc1_b; ep.out_f16 = Hb;
    gemm_ws<2><<<dim3((Ttot + 127) / 128, Dff / 128), 256, 0, stream>>>(
        A16, Wfc1, Ttot, D, ep);
  }

  // 8) FC2 + bias + X1 residual -> d_out (f32)
  {
    EpiParams ep{};
    ep.bias = fc2_b; ep.resid = X1; ep.out_f32 = (float*)d_out;
    gemm_ws<3><<<dim3((Ttot + 127) / 128, D / 128), 256, 0, stream>>>(
        Hb, Wfc2, Ttot, Dff, ep);
  }
}
